// NonAverLoop_16810501996638
// MI455X (gfx1250) — matrix operand order repaired and checked by reading, not compiled, not run on hardware
//
#include <hip/hip_runtime.h>

typedef __attribute__((ext_vector_type(16))) _Float16 v16h;
typedef __attribute__((ext_vector_type(8)))  _Float16 v8h;
typedef __attribute__((ext_vector_type(8)))  float    v8f;

static __device__ __forceinline__ v16h cat8(v8h lo, v8h hi) {
  return __builtin_shufflevector(lo, hi, 0,1,2,3,4,5,6,7,8,9,10,11,12,13,14,15);
}

// A-fragment (M=16,K=32) from row-major [16 x ld] f16 matrix, k-offset k0.
// lane<16: row lane&15, K = k0..k0+15 ; lane>=16: K = k0+16..k0+31
static __device__ __forceinline__ v16h load_fragA(const _Float16* base, int ld, int k0) {
  int lane = threadIdx.x & 31;
  const _Float16* p = base + (lane & 15) * ld + k0 + ((lane >> 4) << 4);
  return cat8(*(const v8h*)p, *(const v8h*)(p + 8));
}

// B-fragment (K=32,N=16) with B[k][n] = rowmajor[n][k] (weights stored [N x ld]).
// lane<16: col lane&15, K = k0..k0+15 ; lane>=16: same col, K = k0+16..k0+31
static __device__ __forceinline__ v16h load_fragB(const _Float16* base, int ld, int k0) {
  int lane = threadIdx.x & 31;
  const _Float16* p = base + (lane & 15) * ld + k0 + ((lane >> 4) << 4);
  return cat8(*(const v8h*)p, *(const v8h*)(p + 8));
}

static __device__ __forceinline__ v8f wmma_f16(v16h a, v16h b, v8f c) {
  return __builtin_amdgcn_wmma_f32_16x16x32_f16(false, a, false, b, (short)0, c,
                                                false, false);
}

static __device__ __forceinline__ float gelu_exact(float v) {
  return 0.5f * v * (1.0f + erff(v * 0.70710678118f));
}

// ---------------------------------------------------------------- converts --
__global__ void cvt_f32_f16_kernel(const float* __restrict__ s,
                                   _Float16* __restrict__ d, int n) {
  int i = blockIdx.x * blockDim.x + threadIdx.x;
  if (i < n) d[i] = (_Float16)s[i];
}

// ------------------------------------------------------------------- stage 1
// One wave per 4x4 patch (16 tokens x 96 ch). 4 waves / 128-thread block.
#define LDP 104   // padded f16 row stride (multiple of 8 halfs = 16B)

__global__ __launch_bounds__(128) void matcore_kernel(
    const float* __restrict__ x, const _Float16* __restrict__ Wmh,
    const float* __restrict__ bm, const _Float16* __restrict__ Wnh,
    const float* __restrict__ bn, float* __restrict__ x1) {
  __shared__ _Float16 sp[4][16 * LDP];   // p tile (f16)
  __shared__ float    slg[4][16 * 96];   // logits, reused as (p+core) out
  __shared__ _Float16 sa[4][16 * LDP];   // softmax(a) (f16)
  __shared__ _Float16 ss[4][16 * 32];    // s tile, K padded to 32

  const int wv = threadIdx.x >> 5, lane = threadIdx.x & 31;
  const int pid = blockIdx.x * 4 + wv;          // 16384 patches
  const int b = pid >> 12, rem = pid & 4095;
  const int ph = rem >> 6, pw = rem & 63;
  const float* xb = x + (size_t)b * 96 * 65536;

  _Float16* P  = sp[wv];
  float*    LG = slg[wv];
  _Float16* Aa = sa[wv];
  _Float16* S  = ss[wv];

  // ---- load patch p[t][c], t = py*4+px ----
  for (int i = lane; i < 16 * 96; i += 32) {
    int c = i >> 4, t = i & 15;
    float v = xb[(size_t)c * 65536 + (ph * 4 + (t >> 2)) * 256 + pw * 4 + (t & 3)];
    P[t * LDP + c] = (_Float16)v;
  }
  __syncthreads();

  // ---- logits = p @ Wm^T + bm ----
  for (int n0 = 0; n0 < 96; n0 += 16) {
    v8f acc = {};
#pragma unroll
    for (int k0 = 0; k0 < 96; k0 += 32) {
      v16h av = load_fragA(P, LDP, k0);
      v16h bv = load_fragB(Wmh + n0 * 96, 96, k0);
      acc = wmma_f16(av, bv, acc);
    }
    int n = n0 + (lane & 15);
    float bias = bm[n];
    int mb = (lane >> 4) * 8;
#pragma unroll
    for (int r = 0; r < 8; ++r) LG[(mb + r) * 96 + n] = acc[r] + bias;
  }
  __syncthreads();

  // ---- softmax over channels per token row ----
  if (lane < 16) {
    const float* row = LG + lane * 96;
    float mx = -1e30f;
    for (int c = 0; c < 96; ++c) mx = fmaxf(mx, row[c]);
    float sum = 0.f;
    for (int c = 0; c < 96; ++c) {
      float e = __expf(row[c] - mx);
      sum += e;
      Aa[lane * LDP + c] = (_Float16)e;
    }
    float inv = 1.0f / sum;
    for (int c = 0; c < 96; ++c)
      Aa[lane * LDP + c] = (_Float16)((float)Aa[lane * LDP + c] * inv);
  }
  __syncthreads();

  // ---- s = a @ p^T  (16x16, K=96) ----
  {
    v8f acc = {};
#pragma unroll
    for (int k0 = 0; k0 < 96; k0 += 32) {
      v16h av = load_fragA(Aa, LDP, k0);
      v16h bv = load_fragB(P, LDP, k0);   // B[k][m] = p[m][k]
      acc = wmma_f16(av, bv, acc);
    }
    int n = lane & 15, mb = (lane >> 4) * 8;
#pragma unroll
    for (int r = 0; r < 8; ++r) S[(mb + r) * 32 + n] = (_Float16)acc[r];
    for (int i = lane; i < 256; i += 32)            // zero-pad K=16..31
      S[(i >> 4) * 32 + 16 + (i & 15)] = (_Float16)0.0f;
  }
  __syncthreads();

  // ---- core = gelu(s @ Wn^T + bn); out = p + core (into LG) ----
  {
    v16h av = load_fragA(S, 32, 0);
    int col = lane & 15, mb = (lane >> 4) * 8;
    for (int n0 = 0; n0 < 96; n0 += 16) {
      const _Float16* wrow = Wnh + (n0 + col) * 16;
      v8h lo = *(const v8h*)wrow;
      v8h hi = *(const v8h*)(wrow + 8);
      v8h z = {};
      if (lane >= 16) { lo = z; hi = z; }           // K=16..31 is zero pad
      v16h bv = cat8(lo, hi);
      v8f acc = {};
      acc = wmma_f16(av, bv, acc);
      int n = n0 + col;
      float bias = bn[n];
#pragma unroll
      for (int r = 0; r < 8; ++r) {
        float v = gelu_exact(acc[r] + bias);
        LG[(mb + r) * 96 + n] = (float)P[(mb + r) * LDP + n] + v;
      }
    }
  }
  __syncthreads();

  // ---- un-patchify: coalesced float4 rows of x1 ----
  float* x1b = x1 + (size_t)b * 96 * 65536;
  for (int j = lane; j < 96 * 4; j += 32) {
    int c = j >> 2, py = j & 3;
    float4 v4;
    v4.x = LG[(py * 4 + 0) * 96 + c];
    v4.y = LG[(py * 4 + 1) * 96 + c];
    v4.z = LG[(py * 4 + 2) * 96 + c];
    v4.w = LG[(py * 4 + 3) * 96 + c];
    *(float4*)(x1b + (size_t)c * 65536 + (ph * 4 + py) * 256 + pw * 4) = v4;
  }
}

// ------------------------------------------------------------------- pooling
__global__ void pool2x_kernel(const float* __restrict__ src,
                              float* __restrict__ dst, int Ho, int Wo, int total) {
  int i = blockIdx.x * blockDim.x + threadIdx.x;
  if (i >= total) return;
  int xo = i % Wo, t = i / Wo;
  int yo = t % Ho, bc = t / Ho;
  const float* s = src + (size_t)bc * (4 * Ho * Wo) + (size_t)(yo * 2) * (Wo * 2) + xo * 2;
  dst[i] = 0.25f * (s[0] + s[1] + s[Wo * 2] + s[Wo * 2 + 1]);
}

// ------------------------------------------------------------------- stage 3
// 256 threads (8 waves) per 32-pixel tile (M = two 16-row WMMA tiles).
// Each weight (B) fragment is loaded once and feeds two WMMAs -> halves L2
// weight traffic vs the 16-pixel version.
// fc1: 8 waves x 3 N-tiles (dual M), fc2: 6 waves x 1 N-tile (dual M), LN.
#define LDF 392   // padded f16 row stride (>=384, multiple of 8 halfs)

__global__ __launch_bounds__(256) void mlp_kernel(
    const float* __restrict__ x1, const float* __restrict__ p2,
    const float* __restrict__ p4, const float* __restrict__ p8,
    const _Float16* __restrict__ W1h, const float* __restrict__ b1,
    const _Float16* __restrict__ W2h, const float* __restrict__ b2,
    const float* __restrict__ gamma, const float* __restrict__ beta,
    float* __restrict__ out) {
  __shared__ _Float16 F[32 * LDF];   // input features (f16), 32 pixels
  __shared__ _Float16 Hh[32 * LDF];  // gelu(fc1) (f16)
  __shared__ float    O[32 * 96];    // fc2 out (f32)
  __shared__ float    smu[32], srs[32];

  const int tid = threadIdx.x;
  const int tile = blockIdx.x;                 // 8192 tiles of 32 pixels
  const int b = tile >> 11, rem = tile & 2047;
  const int y = rem >> 3, x0 = (rem & 7) << 5; // 8 tiles per 256-wide row
  const int pix = tid & 31, grp = tid >> 5;    // 8 groups x 32 pixels
  const int xx = x0 + pix;

  const float* x1b = x1 + (size_t)b * 96 * 65536;
  const float* p2b = p2 + (size_t)b * 96 * 16384;
  const float* p4b = p4 + (size_t)b * 96 * 4096;
  const float* p8b = p8 + (size_t)b * 96 * 1024;

  // ---- build f tile: concat(x1, up2, up4, up8) per pixel, 384 channels ----
#pragma unroll
  for (int i = 0; i < 48; ++i) {
    int c = grp + (i << 3);                    // 0..383
    float v;
    if (c < 96)       v = x1b[(size_t)c * 65536 + y * 256 + xx];
    else if (c < 192) v = p2b[(size_t)(c - 96) * 16384 + (y >> 1) * 128 + (xx >> 1)];
    else if (c < 288) v = p4b[(size_t)(c - 192) * 4096 + (y >> 2) * 64 + (xx >> 2)];
    else              v = p8b[(size_t)(c - 288) * 1024 + (y >> 3) * 32 + (xx >> 3)];
    F[pix * LDF + c] = (_Float16)v;
  }
  __syncthreads();

  const int wv = tid >> 5, lane = tid & 31;

  // ---- fc1 (384 -> 384) + GELU, dual M-tiles per B fragment ----
  for (int j = 0; j < 3; ++j) {
    int n0 = (wv * 3 + j) << 4;
    __builtin_prefetch(W1h + (size_t)(n0 + (lane & 15)) * 384, 0, 1);
    v8f acc0 = {}, acc1 = {};
#pragma unroll
    for (int k0 = 0; k0 < 384; k0 += 32) {
      v16h bv = load_fragB(W1h + (size_t)n0 * 384, 384, k0);
      v16h a0 = load_fragA(F, LDF, k0);
      v16h a1 = load_fragA(F + 16 * LDF, LDF, k0);
      acc0 = wmma_f16(a0, bv, acc0);
      acc1 = wmma_f16(a1, bv, acc1);
    }
    int n = n0 + (lane & 15);
    float bias = b1[n];
    int mb = (lane >> 4) * 8;
#pragma unroll
    for (int r = 0; r < 8; ++r) {
      Hh[(mb + r) * LDF + n]        = (_Float16)gelu_exact(acc0[r] + bias);
      Hh[(16 + mb + r) * LDF + n]   = (_Float16)gelu_exact(acc1[r] + bias);
    }
  }
  __syncthreads();

  // ---- fc2 (384 -> 96), dual M-tiles per B fragment ----
  if (wv < 6) {
    int n0 = wv << 4;
    __builtin_prefetch(W2h + (size_t)(n0 + (lane & 15)) * 384, 0, 1);
    v8f acc0 = {}, acc1 = {};
#pragma unroll
    for (int k0 = 0; k0 < 384; k0 += 32) {
      v16h bv = load_fragB(W2h + (size_t)n0 * 384, 384, k0);
      v16h a0 = load_fragA(Hh, LDF, k0);
      v16h a1 = load_fragA(Hh + 16 * LDF, LDF, k0);
      acc0 = wmma_f16(a0, bv, acc0);
      acc1 = wmma_f16(a1, bv, acc1);
    }
    int n = n0 + (lane & 15);
    float bias = b2[n];
    int mb = (lane >> 4) * 8;
#pragma unroll
    for (int r = 0; r < 8; ++r) {
      O[(mb + r) * 96 + n]      = acc0[r] + bias;
      O[(16 + mb + r) * 96 + n] = acc1[r] + bias;
    }
  }
  __syncthreads();

  // ---- LayerNorm stats per pixel ----
  if (tid < 32) {
    float s = 0.f, s2 = 0.f;
    for (int c = 0; c < 96; ++c) {
      float v = O[tid * 96 + c];
      s += v; s2 += v * v;
    }
    float mu = s * (1.0f / 96.0f);
    float var = s2 * (1.0f / 96.0f) - mu * mu;
    smu[tid] = mu;
    srs[tid] = rsqrtf(var + 1e-5f);
  }
  __syncthreads();

  // ---- normalize + coalesced NCHW store ----
  float* ob = out + (size_t)b * 96 * 65536;
#pragma unroll
  for (int i = 0; i < 12; ++i) {
    int c = grp + (i << 3);                    // 0..95
    float v = (O[pix * 96 + c] - smu[pix]) * srs[pix] * gamma[c] + beta[c];
    ob[(size_t)c * 65536 + y * 256 + xx] = v;
  }
}

// -------------------------------------------------------------------- launch
extern "C" void kernel_launch(void* const* d_in, const int* in_sizes, int n_in,
                              void* d_out, int out_size, void* d_ws, size_t ws_size,
                              hipStream_t stream) {
  const float* x     = (const float*)d_in[0];
  const float* Wm    = (const float*)d_in[1];
  const float* bm    = (const float*)d_in[2];
  const float* Wn    = (const float*)d_in[3];
  const float* bn    = (const float*)d_in[4];
  const float* W1    = (const float*)d_in[5];
  const float* b1    = (const float*)d_in[6];
  const float* W2    = (const float*)d_in[7];
  const float* b2    = (const float*)d_in[8];
  const float* gamma = (const float*)d_in[9];
  const float* beta  = (const float*)d_in[10];
  float* out = (float*)d_out;

  // workspace layout (16B aligned)
  char* ws = (char*)d_ws;
  float*    x1  = (float*)(ws + 0);                       //  96 MB
  float*    p2  = (float*)(ws + 100663296);               //  24 MB
  float*    p4  = (float*)(ws + 125829120);               //   6 MB
  float*    p8  = (float*)(ws + 132120576);               // 1.5 MB
  _Float16* W1h = (_Float16*)(ws + 133693440);            // 288 KB
  _Float16* W2h = (_Float16*)(ws + 133988352);            //  72 KB
  _Float16* Wmh = (_Float16*)(ws + 134062080);            //  18 KB
  _Float16* Wnh = (_Float16*)(ws + 134080512);            //   3 KB

  // 1) weights -> f16
  cvt_f32_f16_kernel<<<(9216 + 255) / 256, 256, 0, stream>>>(Wm, Wmh, 9216);
  cvt_f32_f16_kernel<<<(1536 + 255) / 256, 256, 0, stream>>>(Wn, Wnh, 1536);
  cvt_f32_f16_kernel<<<(147456 + 255) / 256, 256, 0, stream>>>(W1, W1h, 147456);
  cvt_f32_f16_kernel<<<(36864 + 255) / 256, 256, 0, stream>>>(W2, W2h, 36864);

  // 2) MatCore + skip + un-patchify -> x1
  matcore_kernel<<<4096, 128, 0, stream>>>(x, Wmh, bm, Wnh, bn, x1);

  // 3) cascaded 2x average pools
  {
    int t2 = 4 * 96 * 128 * 128;
    pool2x_kernel<<<(t2 + 255) / 256, 256, 0, stream>>>(x1, p2, 128, 128, t2);
    int t4 = 4 * 96 * 64 * 64;
    pool2x_kernel<<<(t4 + 255) / 256, 256, 0, stream>>>(p2, p4, 64, 64, t4);
    int t8 = 4 * 96 * 32 * 32;
    pool2x_kernel<<<(t8 + 255) / 256, 256, 0, stream>>>(p4, p8, 32, 32, t8);
  }

  // 4) fused MultiAver-gather + MLP + LayerNorm (32 pixels / workgroup)
  mlp_kernel<<<8192, 256, 0, stream>>>(x1, p2, p4, p8, W1h, b1, W2h, b2,
                                       gamma, beta, out);
}
